// NaiveLSTM_41463614276079
// MI455X (gfx1250) — compile-verified, matrix operand order checked by reading, not run on hardware
//
#include <hip/hip_runtime.h>
#include <stdint.h>

namespace {
constexpr int LB = 64;          // batch
constexpr int LT = 1024;        // timesteps
constexpr int LI = 512;         // input features
constexpr int LH = 512;         // hidden size
constexpr int LG = 4 * LH;      // 2048 gate columns
constexpr int LK = LI + LH;     // 1024 combined K ([x_t | h_{t-1}])
constexpr int NWG = LH / 16;    // 32 workgroups, one per 16-wide h-column tile
constexpr int WAVES = LB / 16;  // 4 waves per WG, one per 16-row batch tile
constexpr int THREADS = WAVES * 32;
constexpr int KBLKS = LK / 32;      // 32 k-blocks (WMMA K=32)
constexpr int FRAGS = 4 * KBLKS;    // 128 B-fragments resident in LDS
constexpr size_t LDS_BYTES = (size_t)FRAGS * 32 * 32;   // 128 KB
}

typedef __attribute__((ext_vector_type(16))) __bf16 v16bf;
typedef __attribute__((ext_vector_type(8)))  float  v8f;
typedef __attribute__((ext_vector_type(4)))  float  f32x4;   // clang ext_vector for NT load

union Frag32B {
    uint4 u[2];
    v16bf v;
};

static __device__ __forceinline__ unsigned short f2bf(float f) {
    union { float f; unsigned u; } x;
    x.f = f;
    unsigned u = x.u;
    u += 0x7FFFu + ((u >> 16) & 1u);    // round-to-nearest-even
    return (unsigned short)(u >> 16);
}

// ---------------- one-time x -> bf16 conversion (fully coalesced) ------------
// NT load: f32 x is read exactly once -- don't let it occupy L2.
__global__ void cvt_x_bf16(const float* __restrict__ x,
                           unsigned short* __restrict__ xb, int n) {
    int i = (blockIdx.x * blockDim.x + threadIdx.x) * 4;
    if (i + 3 < n) {
        f32x4 v = __builtin_nontemporal_load(reinterpret_cast<const f32x4*>(x + i));
        ushort4 o;
        o.x = f2bf(v.x); o.y = f2bf(v.y); o.z = f2bf(v.z); o.w = f2bf(v.w);
        *reinterpret_cast<ushort4*>(xb + i) = o;   // RT store: keep x_bf16 in L2
    }
}

// ---------------- zero h double-buffer + grid-barrier words ------------------
__global__ void init_state(unsigned short* hb, unsigned* bar) {
    int t = threadIdx.x;
    for (int i = t; i < 2 * LB * LH; i += blockDim.x) hb[i] = 0;
    if (t < 2) bar[t] = 0;
}

// ---------------- grid-wide barrier (32 co-resident WGs) ---------------------
static __device__ __forceinline__ void grid_barrier(unsigned* cnt,
                                                    volatile unsigned* gen) {
    __threadfence();          // make this step's h stores agent-visible
    __syncthreads();
    if (threadIdx.x == 0) {
        unsigned g = *gen;
        unsigned old = atomicAdd(cnt, 1u);
        if (old == NWG - 1) {
            *cnt = 0;
            __threadfence();
            atomicAdd((unsigned*)gen, 1u);
        } else {
            while (*gen == g) { __builtin_amdgcn_s_sleep(1); }
        }
    }
    __threadfence();          // acquire side before reading new h
    __syncthreads();
}

// ---------------- persistent fused LSTM kernel -------------------------------
// wave (wv, blockIdx.x=hc) owns batch rows [wv*16,wv*16+16) x h cols [hc*16,hc*16+16)
// gates_t = [x_t | h_{t-1}] (64x1024 bf16)  @  [W;U] (1024x2048 bf16, in LDS)  + bias
__launch_bounds__(THREADS, 1)
__global__ void lstm_persistent(const unsigned short* __restrict__ xb,
                                unsigned short* __restrict__ hb,   // 2 x (64x512) bf16
                                const float* __restrict__ W,
                                const float* __restrict__ U,
                                const float* __restrict__ bias,
                                float* __restrict__ out,
                                unsigned* __restrict__ bar) {
    extern __shared__ unsigned char smem[];
    // LDS layout: [kb 0..31][gate 0..3][lane 0..31][32B]  -> one kb spans 4KB,
    // so all 8 ds_load_b128 of a kb-iteration use immediate offsets.

    const int tid  = threadIdx.x;
    const int lane = tid & 31;
    const int wv   = tid >> 5;          // batch tile 0..3
    const int hc   = blockIdx.x;        // h tile 0..31
    const int sub  = lane & 15;
    const int hi   = lane >> 4;         // which K-half of the fragment

    // ---- one-time: pre-swizzle [W;U] columns for this WG into B-fragments ----
    // fragment f = kb*4 + q ; lane l holds 16 bf16: element j has
    //   K = kb*32 + (l>>4)*8 + (j<8 ? j : j+8),  N = q*512 + hc*16 + (l&15)
    for (int idx = tid; idx < FRAGS * 32; idx += THREADS) {
        int f = idx >> 5, l = idx & 31;
        int kb = f >> 2, q = f & 3;
        int lhi = l >> 4, lsub = l & 15;
        int col = q * LH + hc * 16 + lsub;
        unsigned short tmp[16];
#pragma unroll
        for (int j = 0; j < 16; ++j) {
            int kk = kb * 32 + lhi * 8 + (j < 8 ? j : j + 8);
            float v = (kk < LI) ? W[(size_t)kk * LG + col]
                                : U[(size_t)(kk - LI) * LG + col];
            tmp[j] = f2bf(v);
        }
        uint4* dst = reinterpret_cast<uint4*>(smem + (size_t)idx * 32);
        dst[0] = *reinterpret_cast<uint4*>(tmp);
        dst[1] = *reinterpret_cast<uint4*>(tmp + 8);
    }
    __syncthreads();

    // per-lane LDS base: fragment data for this lane starts at lane*32
    const unsigned char* lds_lane = smem + (size_t)lane * 32;

    // bias for this lane's output column, per gate (C/D layout: N = lane&15)
    const int col = hc * 16 + sub;
    float bq[4];
#pragma unroll
    for (int q = 0; q < 4; ++q) bq[q] = bias[q * LH + col];

    // cell state lives in registers for the whole sequence
    v8f c = {0.f, 0.f, 0.f, 0.f, 0.f, 0.f, 0.f, 0.f};

    const int rowbase = wv * 16;
    const int bA = rowbase + sub;       // A-fragment row for this lane (both K-halves)

    for (int t = 0; t < LT; ++t) {
        const unsigned short* __restrict__ hrd = hb + (size_t)(t & 1) * LB * LH;
        unsigned short* __restrict__ hwr       = hb + (size_t)((t + 1) & 1) * LB * LH;

        v8f acc[4];
#pragma unroll
        for (int q = 0; q < 4; ++q)
            acc[q] = (v8f){0.f, 0.f, 0.f, 0.f, 0.f, 0.f, 0.f, 0.f};

        const unsigned short* __restrict__ xrow = xb + ((size_t)bA * LT + t) * LI;
        const unsigned short* __restrict__ hrow = hrd + (size_t)bA * LH;

#pragma unroll 4
        for (int kb = 0; kb < KBLKS; ++kb) {
            // A fragment: 16x32 bf16, row-major memory == fragment layout
            const unsigned short* src;
            int off;
            if (kb < LI / 32) { src = xrow; off = kb * 32 + hi * 8; }
            else              { src = hrow; off = (kb - LI / 32) * 32 + hi * 8; }
            Frag32B a;
            a.u[0] = *reinterpret_cast<const uint4*>(src + off);
            a.u[1] = *reinterpret_cast<const uint4*>(src + off + 16);

#pragma unroll
            for (int q = 0; q < 4; ++q) {
                Frag32B b;
                const uint4* bp = reinterpret_cast<const uint4*>(
                    lds_lane + (size_t)(kb * 4 + q) * 1024);
                b.u[0] = bp[0];
                b.u[1] = bp[1];
                acc[q] = __builtin_amdgcn_wmma_f32_16x16x32_bf16(
                    false, a.v, false, b.v, (short)0, acc[q], false, false);
            }
        }

        // elementwise gate math; C/D layout: vgpr v -> batch row rowbase+v+8*hi
        float ht[8];
#pragma unroll
        for (int v = 0; v < 8; ++v) {
            float gi = acc[0][v] + bq[0];
            float gf = acc[1][v] + bq[1];
            float gg = acc[2][v] + bq[2];
            float go = acc[3][v] + bq[3];
            float it = 1.f / (1.f + __expf(-gi));
            float ft = 1.f / (1.f + __expf(-gf));
            float gt = tanhf(gg);
            float ot = 1.f / (1.f + __expf(-go));
            float cn = ft * c[v] + it * gt;
            c[v] = cn;
            ht[v] = ot * tanhf(cn);
        }

#pragma unroll
        for (int v = 0; v < 8; ++v) {
            int b = rowbase + v + hi * 8;
            // hidden_seq is write-once: NT store keeps x_bf16/h resident in L2
            __builtin_nontemporal_store(ht[v], out + ((size_t)b * LT + t) * LH + col);
            hwr[(size_t)b * LH + col] = f2bf(ht[v]);             // next-step h (RT)
        }
        if (t == LT - 1) {
            size_t base = (size_t)LB * LT * LH;
#pragma unroll
            for (int v = 0; v < 8; ++v) {
                int b = rowbase + v + hi * 8;
                out[base + (size_t)b * LH + col] = ht[v];                    // h_T
                out[base + (size_t)LB * LH + (size_t)b * LH + col] = c[v];   // c_T
            }
        }

        grid_barrier(bar, bar + 1);
    }
}

// ---------------------------------------------------------------------------
extern "C" void kernel_launch(void* const* d_in, const int* in_sizes, int n_in,
                              void* d_out, int out_size, void* d_ws, size_t ws_size,
                              hipStream_t stream) {
    const float* x    = (const float*)d_in[0];
    const float* W    = (const float*)d_in[1];
    const float* U    = (const float*)d_in[2];
    const float* bias = (const float*)d_in[3];
    float* out        = (float*)d_out;

    // workspace: [ x_bf16 : B*T*I bf16 ][ h double-buffer : 2*B*H bf16 ][ barrier ]
    unsigned short* xb = (unsigned short*)d_ws;
    size_t xbytes = (size_t)LB * LT * LI * sizeof(unsigned short);
    unsigned short* hb = (unsigned short*)((char*)d_ws + xbytes);
    size_t hbytes = (size_t)2 * LB * LH * sizeof(unsigned short);
    unsigned* bar = (unsigned*)((char*)d_ws + xbytes + hbytes);

    int n = LB * LT * LI;
    cvt_x_bf16<<<n / (4 * 256), 256, 0, stream>>>(x, xb, n);
    init_state<<<1, 256, 0, stream>>>(hb, bar);
    lstm_persistent<<<NWG, THREADS, LDS_BYTES, stream>>>(xb, hb, W, U, bias, out, bar);
}
